// MixedLinear_KV_86139864088894
// MI455X (gfx1250) — compile-verified, matrix operand order checked by reading, not run on hardware
//
#include <hip/hip_runtime.h>
#include <hip/hip_bf16.h>
#include <stdint.h>
#include <stddef.h>

typedef __attribute__((ext_vector_type(16))) __bf16 v16bf;
typedef __attribute__((ext_vector_type(8)))  __bf16 v8bf;
typedef __attribute__((ext_vector_type(4)))  __bf16 v4bf;
typedef __attribute__((ext_vector_type(8)))  float  v8f;
typedef __attribute__((ext_vector_type(4)))  float  v4f;
typedef __attribute__((ext_vector_type(4)))  int    v4i;

#define M_TOT 32768   // B*S = 8*4096
#define K_TOT 1024    // MAX_IN
#define N_TOT 512     // MAX_OUT

// ---------------- CDNA5 async global->LDS (guarded) ----------------
#if defined(__has_builtin)
# if __has_builtin(__builtin_amdgcn_global_load_async_to_lds_b128)
#  define HAVE_ASYNC_LDS 1
# endif
# if __has_builtin(__builtin_amdgcn_s_wait_asynccnt)
#  define HAVE_WAIT_ASYNC 1
# endif
#endif

#if HAVE_ASYNC_LDS
typedef __attribute__((address_space(1))) v4i GV4;   // global
typedef __attribute__((address_space(3))) v4i LV4;   // LDS
#endif

__device__ __forceinline__ void async_copy16(const void* g, void* l) {
#if HAVE_ASYNC_LDS
  __builtin_amdgcn_global_load_async_to_lds_b128((GV4*)g, (LV4*)l, 0, 0);
#else
  *(v8bf*)l = *(const v8bf*)g;   // global_load_b128 + ds_store_b128 fallback
#endif
}

__device__ __forceinline__ void wait_async_le4() {
#if HAVE_ASYNC_LDS
# if HAVE_WAIT_ASYNC
  __builtin_amdgcn_s_wait_asynccnt(4);
# else
  asm volatile("s_wait_asynccnt 4" ::: "memory");
# endif
#endif
}

__device__ __forceinline__ void wait_async_le0() {
#if HAVE_ASYNC_LDS
# if HAVE_WAIT_ASYNC
  __builtin_amdgcn_s_wait_asynccnt(0);
# else
  asm volatile("s_wait_asynccnt 0" ::: "memory");
# endif
#endif
}

// ---------------- Prep: w_mix (bf16, [N_TOT][K_TOT]) + b_mix ----------------
__global__ __launch_bounds__(256)
void mlkv_prep_w(const float* __restrict__ weight,   // [512][1024]
                 const float* __restrict__ bias,     // [512]
                 const float* __restrict__ mw,       // [36]
                 const float* __restrict__ w_scales, // [2]
                 __bf16* __restrict__ wq,            // [512][1024] bf16
                 float*  __restrict__ bmix) {        // [512]
  const int idx = blockIdx.x * 256 + threadIdx.x;
  if (idx >= N_TOT * K_TOT) return;
  const int o = idx >> 10;     // out index
  const int h = idx & 1023;    // in index

  const int HS[3] = {512, 768, 1024};
  const int NH[3] = {8, 12, 16};

  // k = (i*3+j)*4 + m*2 + n  (m = a-bit idx, n = w-bit idx)
  float c4 = 0.f, c8 = 0.f;
#pragma unroll
  for (int i = 0; i < 3; ++i) {
#pragma unroll
    for (int j = 0; j < 3; ++j) {
      const int od = 4 * (HS[i] / NH[j]);
      if (h < HS[i] && o < od) {
        const int k0 = (i * 3 + j) * 4;
        c4 += mw[k0 + 0] + mw[k0 + 2];   // n=0, m={0,1}
        c8 += mw[k0 + 1] + mw[k0 + 3];   // n=1, m={0,1}
      }
    }
  }
  const float s4 = w_scales[0], s8 = w_scales[1];
  const float w  = weight[idx];
  const float q4 = rintf(fminf(fmaxf(w / s4,   -8.f),   7.f));
  const float q8 = rintf(fminf(fmaxf(w / s8, -128.f), 127.f));
  wq[idx] = (__bf16)(c4 * s4 * q4 + c8 * s8 * q8);

  if (h == 0) {   // bias padding depends only on o
    float cb = 0.f;
#pragma unroll
    for (int i = 0; i < 3; ++i) {
#pragma unroll
      for (int j = 0; j < 3; ++j) {
        const int od = 4 * (HS[i] / NH[j]);
        if (o < od) {
          const int k0 = (i * 3 + j) * 4;
          cb += mw[k0] + mw[k0 + 1] + mw[k0 + 2] + mw[k0 + 3];
        }
      }
    }
    bmix[o] = cb * bias[o];
  }
}

// ---------------- Prep: x_mix (bf16, [M_TOT][K_TOT]) ----------------
__global__ __launch_bounds__(256)
void mlkv_prep_x(const float* __restrict__ x,        // [M_TOT][K_TOT]
                 const float* __restrict__ mw,       // [36]
                 const float* __restrict__ a_scales, // [2]
                 __bf16* __restrict__ xq) {          // [M_TOT][K_TOT]
  const size_t q = (size_t)blockIdx.x * 256 + threadIdx.x;  // unit of 4 floats
  float ca0 = 0.f, ca1 = 0.f;
#pragma unroll
  for (int k = 0; k < 36; k += 4) {  // k%4 = m*2+n  -> m=0:{0,1}, m=1:{2,3}
    ca0 += mw[k + 0] + mw[k + 1];
    ca1 += mw[k + 2] + mw[k + 3];
  }
  const float s0 = a_scales[0], s1 = a_scales[1];
  const v4f xv = *(const v4f*)(x + q * 4);
  v4bf r;
#pragma unroll
  for (int e = 0; e < 4; ++e) {
    const float q4 = rintf(fminf(fmaxf(xv[e] / s0,   -8.f),   7.f));
    const float q8 = rintf(fminf(fmaxf(xv[e] / s1, -128.f), 127.f));
    r[e] = (__bf16)(ca0 * s0 * q4 + ca1 * s1 * q8);
  }
  *(v4bf*)(xq + q * 4) = r;
}

// ---------------- GEMM: out = xq @ wq^T + bmix ----------------
#define BM 128
#define BN 128
#define BK 32
#define LDSS 40                 // bf16 elems/row: 32 + 8 pad -> 80B rows (16B-aligned)
#define NSTAGE (K_TOT / BK)     // 32

__global__ __launch_bounds__(256)
void mlkv_gemm(const __bf16* __restrict__ xq,   // [M_TOT][K_TOT]
               const __bf16* __restrict__ wq,   // [N_TOT][K_TOT]
               const float*  __restrict__ bmix, // [N_TOT]
               float* __restrict__ out) {       // [M_TOT][N_TOT]
  __shared__ __bf16 ldsA[2][BM * LDSS];
  __shared__ __bf16 ldsB[2][BN * LDSS];

  const int t   = threadIdx.x;
  const int bm0 = blockIdx.y * BM;
  const int bn0 = blockIdx.x * BN;

  auto copy_stage = [&](int s, int buf) {
    const int k0 = s * BK;
#pragma unroll
    for (int i = 0; i < 2; ++i) {
      const int c   = t + i * 256;   // 512 chunks of 16B per tile
      const int row = c >> 2;
      const int cc  = c & 3;
      async_copy16(xq + (size_t)(bm0 + row) * K_TOT + k0 + cc * 8,
                   &ldsA[buf][row * LDSS + cc * 8]);
      async_copy16(wq + (size_t)(bn0 + row) * K_TOT + k0 + cc * 8,
                   &ldsB[buf][row * LDSS + cc * 8]);
    }
  };

  const int lane = t & 31;
  const int wv   = t >> 5;
  const int m0   = (wv & 3) * 32;    // wave's M offset in tile (2 sub-tiles)
  const int n0   = (wv >> 2) * 64;   // wave's N offset in tile (4 sub-tiles)
  const int r    = lane & 15;
  const int hi   = lane >> 4;
  const int foff = hi * 8;           // K{8..15} for upper half-wave

  v8f acc[2][4];
#pragma unroll
  for (int a = 0; a < 2; ++a)
#pragma unroll
    for (int b = 0; b < 4; ++b)
      acc[a][b] = {};

  copy_stage(0, 0);

  for (int s = 0; s < NSTAGE; ++s) {
    const int buf = s & 1;
    if (s + 1 < NSTAGE) { copy_stage(s + 1, buf ^ 1); wait_async_le4(); }
    else                { wait_async_le0(); }
    __syncthreads();

    v16bf afrag[2];
#pragma unroll
    for (int a = 0; a < 2; ++a) {
      const __bf16* base = &ldsA[buf][(m0 + a * 16 + r) * LDSS + foff];
      v8bf lo = *(const v8bf*)(base);          // K 0..7   (or 8..15)
      v8bf hv = *(const v8bf*)(base + 16);     // K 16..23 (or 24..31)
      afrag[a] = __builtin_shufflevector(lo, hv,
                   0,1,2,3,4,5,6,7,8,9,10,11,12,13,14,15);
    }
    v16bf bfrag[4];
#pragma unroll
    for (int b = 0; b < 4; ++b) {
      const __bf16* base = &ldsB[buf][(n0 + b * 16 + r) * LDSS + foff];
      v8bf lo = *(const v8bf*)(base);
      v8bf hv = *(const v8bf*)(base + 16);
      bfrag[b] = __builtin_shufflevector(lo, hv,
                   0,1,2,3,4,5,6,7,8,9,10,11,12,13,14,15);
    }

#pragma unroll
    for (int a = 0; a < 2; ++a)
#pragma unroll
      for (int b = 0; b < 4; ++b)
        acc[a][b] = __builtin_amdgcn_wmma_f32_16x16x32_bf16(
            false, afrag[a], false, bfrag[b],
            (short)0, acc[a][b], false, false);

    __syncthreads();
  }

  // Epilogue: C/D layout — VGPR v, lanes<16 -> M=v, lanes>=16 -> M=v+8; N=lane&15
#pragma unroll
  for (int b = 0; b < 4; ++b) {
    const int gn   = bn0 + n0 + b * 16 + r;
    const float bb = bmix[gn];
#pragma unroll
    for (int a = 0; a < 2; ++a) {
#pragma unroll
      for (int v = 0; v < 8; ++v) {
        const int gm = bm0 + m0 + a * 16 + hi * 8 + v;
        out[(size_t)gm * N_TOT + gn] = acc[a][b][v] + bb;
      }
    }
  }
}

// ---------------- Host launcher ----------------
extern "C" void kernel_launch(void* const* d_in, const int* in_sizes, int n_in,
                              void* d_out, int out_size, void* d_ws, size_t ws_size,
                              hipStream_t stream) {
  (void)in_sizes; (void)n_in; (void)out_size; (void)ws_size;
  const float* x        = (const float*)d_in[0];  // [8,4096,1024]
  const float* weight   = (const float*)d_in[1];  // [512,1024]
  const float* bias     = (const float*)d_in[2];  // [512]
  const float* mix_w    = (const float*)d_in[3];  // [36]
  const float* a_scales = (const float*)d_in[4];  // [2]
  const float* w_scales = (const float*)d_in[5];  // [2]
  float* out            = (float*)d_out;          // [8,4096,512]

  // Workspace layout
  char* ws = (char*)d_ws;
  __bf16* xq   = (__bf16*)(ws);                                      // 64 MB
  __bf16* wqp  = (__bf16*)(ws + (size_t)M_TOT * K_TOT * 2);          // 1 MB
  float*  bmix = (float*) (ws + (size_t)M_TOT * K_TOT * 2
                              + (size_t)N_TOT * K_TOT * 2);          // 2 KB

  mlkv_prep_w<<<(N_TOT * K_TOT) / 256, 256, 0, stream>>>(
      weight, bias, mix_w, w_scales, wqp, bmix);
  mlkv_prep_x<<<(int)(((size_t)M_TOT * K_TOT / 4) / 256), 256, 0, stream>>>(
      x, mix_w, a_scales, xq);
  mlkv_gemm<<<dim3(N_TOT / BN, M_TOT / BM), 256, 0, stream>>>(
      xq, wqp, bmix, out);
}
